// WarpIndexEpipolarAttention_57561151701643
// MI455X (gfx1250) — matrix operand order repaired and checked by reading, not compiled, not run on hardware
//
#include <hip/hip_runtime.h>

// ---------------------------------------------------------------------------
// Types
// ---------------------------------------------------------------------------
typedef __bf16 bf16_t;
typedef bf16_t bf16x16 __attribute__((ext_vector_type(16)));
typedef bf16_t bf16x8  __attribute__((ext_vector_type(8)));
typedef bf16_t bf16x4  __attribute__((ext_vector_type(4)));
typedef float  f32x8   __attribute__((ext_vector_type(8)));

#define M_ROWS   16384   // B * HW = 4 * 4096
#define HWD      4096
#define CDIM     128
#define HIDDIM   256
#define DCAND    64
#define EPSA     1e-10f

// ---------------------------------------------------------------------------
// Helpers
// ---------------------------------------------------------------------------
__device__ __forceinline__ float wred_add(float v) {
#pragma unroll
  for (int o = 16; o > 0; o >>= 1) v += __shfl_xor(v, o, 32);
  return v;
}
__device__ __forceinline__ float wred_max(float v) {
#pragma unroll
  for (int o = 16; o > 0; o >>= 1) v = fmaxf(v, __shfl_xor(v, o, 32));
  return v;
}
__device__ __forceinline__ float gelu_exact(float x) {
  return 0.5f * x * (1.0f + erff(x * 0.7071067811865476f));
}

// ---------------------------------------------------------------------------
// Weight prep: dst[n*K + k] = (bf16) src[k*N + n]   (N-major for B-fragments)
// ---------------------------------------------------------------------------
__global__ void wprep_kernel(const float* __restrict__ src, bf16_t* __restrict__ dst,
                             int K, int N) {
  int t = blockIdx.x * 256 + threadIdx.x;
  if (t >= K * N) return;
  int n = t / K, k = t % K;
  dst[t] = (bf16_t)src[k * N + n];
}

// ---------------------------------------------------------------------------
// LN1 + transpose: x[m][ch] = feat[(nB*128+ch)*4096 + i]; write x fp32 + LN bf16
// one wave per row, block = 8 waves
// ---------------------------------------------------------------------------
__global__ void ln1_transpose_kernel(const float* __restrict__ feat,
                                     const float* __restrict__ g, const float* __restrict__ b,
                                     float* __restrict__ xout, bf16_t* __restrict__ xnb) {
  int w = threadIdx.x >> 5, lane = threadIdx.x & 31;
  int m = blockIdx.x * 8 + w;
  int nB = m >> 12, i = m & 4095;
  const float* fb = feat + (size_t)nB * 128 * HWD + i;
  float v0 = fb[(size_t)(lane)      * HWD];
  float v1 = fb[(size_t)(lane + 32) * HWD];
  float v2 = fb[(size_t)(lane + 64) * HWD];
  float v3 = fb[(size_t)(lane + 96) * HWD];
  float mean = wred_add(v0 + v1 + v2 + v3) * (1.0f / 128.0f);
  float d0 = v0 - mean, d1 = v1 - mean, d2 = v2 - mean, d3 = v3 - mean;
  float var = wred_add(d0 * d0 + d1 * d1 + d2 * d2 + d3 * d3) * (1.0f / 128.0f);
  float rstd = rsqrtf(var + 1e-5f);
  float* xo = xout + (size_t)m * CDIM;
  bf16_t* xn = xnb + (size_t)m * CDIM;
  xo[lane] = v0; xo[lane + 32] = v1; xo[lane + 64] = v2; xo[lane + 96] = v3;
  xn[lane]      = (bf16_t)(d0 * rstd * g[lane]      + b[lane]);
  xn[lane + 32] = (bf16_t)(d1 * rstd * g[lane + 32] + b[lane + 32]);
  xn[lane + 64] = (bf16_t)(d2 * rstd * g[lane + 64] + b[lane + 64]);
  xn[lane + 96] = (bf16_t)(d3 * rstd * g[lane + 96] + b[lane + 96]);
}

// ---------------------------------------------------------------------------
// LN over contiguous fp32 rows -> bf16 rows (for LN2)
// ---------------------------------------------------------------------------
__global__ void ln_rows_kernel(const float* __restrict__ src,
                               const float* __restrict__ g, const float* __restrict__ b,
                               bf16_t* __restrict__ dst) {
  int w = threadIdx.x >> 5, lane = threadIdx.x & 31;
  int m = blockIdx.x * 8 + w;
  const float* sb = src + (size_t)m * CDIM;
  float v0 = sb[lane], v1 = sb[lane + 32], v2 = sb[lane + 64], v3 = sb[lane + 96];
  float mean = wred_add(v0 + v1 + v2 + v3) * (1.0f / 128.0f);
  float d0 = v0 - mean, d1 = v1 - mean, d2 = v2 - mean, d3 = v3 - mean;
  float var = wred_add(d0 * d0 + d1 * d1 + d2 * d2 + d3 * d3) * (1.0f / 128.0f);
  float rstd = rsqrtf(var + 1e-5f);
  bf16_t* dn = dst + (size_t)m * CDIM;
  dn[lane]      = (bf16_t)(d0 * rstd * g[lane]      + b[lane]);
  dn[lane + 32] = (bf16_t)(d1 * rstd * g[lane + 32] + b[lane + 32]);
  dn[lane + 64] = (bf16_t)(d2 * rstd * g[lane + 64] + b[lane + 64]);
  dn[lane + 96] = (bf16_t)(d3 * rstd * g[lane + 96] + b[lane + 96]);
}

// ---------------------------------------------------------------------------
// Generic WMMA GEMM: out[m][n] = epi( A[m][:] @ W[:][n] )  (A bf16 row-major,
// Bt = weights bf16 N-major so B-fragment loads are contiguous).
// block = 128 threads (4 waves), each wave -> one 16x16 tile.
// epilogue: v = (acc + bias[n]) * scale; v += addres[m][n]; v = gelu(v) if act
// ---------------------------------------------------------------------------
__global__ void gemm_wmma_kernel(const bf16_t* __restrict__ A, const bf16_t* __restrict__ Bt,
                                 const float* __restrict__ bias, const float* __restrict__ addres,
                                 float* __restrict__ outf, int K, int N, float scale, int act) {
  int wv = threadIdx.x >> 5, lane = threadIdx.x & 31;
  int tileM = blockIdx.x * 16;
  int tileN = (blockIdx.y * 4 + wv) * 16;
  if (tileN >= N) return;
  int mr = lane & 15, half = lane >> 4;

  // A fragment (16x32 bf16): lanes 0-15 -> K = kb + half*8 + {0..7, 16..23}
  const bf16_t* ap = A  + (size_t)(tileM + mr) * K + half * 8;
  // B fragment (32x16 bf16) from N-major weights: K = kb + half*8 + {0..7, 16..23}
  const bf16_t* bp = Bt + (size_t)(tileN + mr) * K + half * 8;

  f32x8 acc = {0.f, 0.f, 0.f, 0.f, 0.f, 0.f, 0.f, 0.f};
  for (int kb = 0; kb < K; kb += 32) {
    union { bf16x16 v; bf16x8 h[2]; } a, bm;
    a.h[0]  = *(const bf16x8*)(ap + kb);
    a.h[1]  = *(const bf16x8*)(ap + kb + 16);
    bm.h[0] = *(const bf16x8*)(bp + kb);
    bm.h[1] = *(const bf16x8*)(bp + kb + 16);
    acc = __builtin_amdgcn_wmma_f32_16x16x32_bf16(false, a.v, false, bm.v,
                                                  (short)0, acc, false, false);
  }

  int col = tileN + mr;
  float bc = bias[col];
  size_t base = (size_t)(tileM + half * 8) * N + col;
#pragma unroll
  for (int r = 0; r < 8; r++) {
    float v = (acc[r] + bc) * scale;
    if (addres) v += addres[base + (size_t)r * N];
    if (act) v = gelu_exact(v);
    outf[base + (size_t)r * N] = v;
  }
}

// ---------------------------------------------------------------------------
// Gather attention. block = 256 (8 waves), one wave per query row m.
// Phase 1: QmK dots (lane owns candidates k=lane, k=lane+32), softmax,
//          depth_prob modulation + renorm.
// Phase 2: AmV accumulation (lane owns 4 channels), attn/idx staged in LDS.
// ---------------------------------------------------------------------------
__global__ void attn_kernel(const float* __restrict__ qb, const float* __restrict__ kb,
                            const float* __restrict__ vb, const int* __restrict__ widx,
                            const float* __restrict__ dp, const float* __restrict__ k_pe,
                            const float* __restrict__ v_pe, const float* __restrict__ rpe,
                            bf16_t* __restrict__ aout) {
  __shared__ __align__(16) float qs[8][CDIM];
  __shared__ float sattn[8][DCAND];
  __shared__ int   sidx[8][DCAND];

  int w = threadIdx.x >> 5, lane = threadIdx.x & 31;
  int m = blockIdx.x * 8 + w;
  int nB = m >> 12, npair = nB ^ 1;
  const float4* K4 = (const float4*)(kb + (size_t)npair * HWD * CDIM);
  const float4* V4 = (const float4*)(vb + (size_t)npair * HWD * CDIM);

  // stage Q row (already scaled by c^-0.5) and indices
  float4 q4 = *(const float4*)(qb + (size_t)m * CDIM + lane * 4);
  *(float4*)&qs[w][lane * 4] = q4;
  float qpe = wred_add(q4.x + q4.y + q4.z + q4.w);   // Q_pe = sum_c Q
  int i0 = widx[(size_t)m * DCAND + lane];
  int i1 = widx[(size_t)m * DCAND + 32 + lane];
  sidx[w][lane] = i0; sidx[w][lane + 32] = i1;
  __syncthreads();

  // cost[k] = Q . K[idx[k]]
  const float4* q4p = (const float4*)&qs[w][0];
  float c0 = 0.f, c1 = 0.f;
  size_t b0 = (size_t)i0 * 32, b1 = (size_t)i1 * 32;
#pragma unroll 4
  for (int c4 = 0; c4 < 32; c4++) {
    float4 qq = q4p[c4];
    float4 ka = K4[b0 + c4];
    float4 kc = K4[b1 + c4];
    c0 += qq.x * ka.x + qq.y * ka.y + qq.z * ka.z + qq.w * ka.w;
    c1 += qq.x * kc.x + qq.y * kc.y + qq.z * kc.z + qq.w * kc.w;
  }
  c0 += qpe * k_pe[lane]      + rpe[lane];
  c1 += qpe * k_pe[lane + 32] + rpe[lane + 32];

  // softmax over 64 candidates
  float mx = wred_max(fmaxf(c0, c1));
  float e0 = __expf(c0 - mx), e1 = __expf(c1 - mx);
  float s = wred_add(e0 + e1);
  float a0 = (e0 / s) * dp[(size_t)m * DCAND + lane];
  float a1 = (e1 / s) * dp[(size_t)m * DCAND + 32 + lane];
  float s2 = wred_add(a0 + a1);
  a0 = (a0 + EPSA) / (s2 + EPSA);
  a1 = (a1 + EPSA) / (s2 + EPSA);
  sattn[w][lane] = a0; sattn[w][lane + 32] = a1;
  float ape = wred_add(a0 * v_pe[lane] + a1 * v_pe[lane + 32]);  // (attn*v_pe).sum
  __syncthreads();

  // out[c] = sum_k attn[k] * V[idx[k]][c]  (lane owns channels lane*4..+3)
  float4 acc = {0.f, 0.f, 0.f, 0.f};
#pragma unroll 4
  for (int k = 0; k < DCAND; k++) {
    float a = sattn[w][k];
    float4 vv = V4[(size_t)sidx[w][k] * 32 + lane];
    acc.x += a * vv.x; acc.y += a * vv.y; acc.z += a * vv.z; acc.w += a * vv.w;
  }
  acc.x += ape; acc.y += ape; acc.z += ape; acc.w += ape;

  bf16x4 o;
  o[0] = (bf16_t)acc.x; o[1] = (bf16_t)acc.y; o[2] = (bf16_t)acc.z; o[3] = (bf16_t)acc.w;
  *(bf16x4*)(aout + (size_t)m * CDIM + lane * 4) = o;
}

// ---------------------------------------------------------------------------
// Depthwise 5x5 conv (SAME, zero pad) on y viewed as (nB, hw, hid); lane = ch.
// z = y + gelu(conv(y) + dw_b)  ->  bf16 for fc2
// ---------------------------------------------------------------------------
__global__ void dwconv_kernel(const float* __restrict__ y, const float* __restrict__ dww,
                              const float* __restrict__ dwb, bf16_t* __restrict__ zb) {
  int ch = threadIdx.x;                // 256
  int m = blockIdx.x;                  // 16384
  int nB = m >> 12, i = m & 4095, h = i >> 6, wp = i & 63;
  const float* ybase = y + (size_t)nB * HWD * HIDDIM + ch;
  const float* wt = dww + ch * 25;
  float acc = 0.f;
#pragma unroll
  for (int dy = 0; dy < 5; dy++) {
    int hh = h + dy - 2;
    if (hh < 0 || hh > 63) continue;
#pragma unroll
    for (int dx = 0; dx < 5; dx++) {
      int ww = wp + dx - 2;
      if (ww < 0 || ww > 63) continue;
      acc += ybase[(size_t)(hh * 64 + ww) * HIDDIM] * wt[dy * 5 + dx];
    }
  }
  float z = y[(size_t)m * HIDDIM + ch] + gelu_exact(acc + dwb[ch]);
  zb[(size_t)m * HIDDIM + ch] = (bf16_t)z;
}

// ---------------------------------------------------------------------------
// Launch
// ---------------------------------------------------------------------------
extern "C" void kernel_launch(void* const* d_in, const int* in_sizes, int n_in,
                              void* d_out, int out_size, void* d_ws, size_t ws_size,
                              hipStream_t stream) {
  (void)in_sizes; (void)n_in; (void)out_size; (void)ws_size;
  const float* feat   = (const float*)d_in[0];
  const int*   widx   = (const int*)  d_in[1];
  const float* dp     = (const float*)d_in[2];
  const float* ln1_g  = (const float*)d_in[3];
  const float* ln1_b  = (const float*)d_in[4];
  const float* wq     = (const float*)d_in[5];
  const float* bq     = (const float*)d_in[6];
  const float* wk     = (const float*)d_in[7];
  const float* bk     = (const float*)d_in[8];
  const float* wv     = (const float*)d_in[9];
  const float* bv     = (const float*)d_in[10];
  const float* k_pe   = (const float*)d_in[11];
  const float* v_pe   = (const float*)d_in[12];
  const float* rpe    = (const float*)d_in[13];
  const float* proj_w = (const float*)d_in[14];
  const float* proj_b = (const float*)d_in[15];
  const float* ln2_g  = (const float*)d_in[16];
  const float* ln2_b  = (const float*)d_in[17];
  const float* fc1_w  = (const float*)d_in[18];
  const float* fc1_b  = (const float*)d_in[19];
  const float* dw_w   = (const float*)d_in[20];
  const float* dw_b   = (const float*)d_in[21];
  const float* fc2_w  = (const float*)d_in[22];
  const float* fc2_b  = (const float*)d_in[23];
  float* out = (float*)d_out;

  char* ws = (char*)d_ws;
  size_t off = 0;
  auto alloc = [&](size_t bytes) -> void* {
    void* p = ws + off;
    off = (off + bytes + 255) & ~(size_t)255;
    return p;
  };

  float*  xbuf  = (float*) alloc((size_t)M_ROWS * CDIM * 4);     // shortcut x
  bf16_t* xnb   = (bf16_t*)alloc((size_t)M_ROWS * CDIM * 2);     // LN1(x) bf16
  float*  qbuf  = (float*) alloc((size_t)M_ROWS * CDIM * 4);
  float*  kbuf  = (float*) alloc((size_t)M_ROWS * CDIM * 4);
  float*  vbuf  = (float*) alloc((size_t)M_ROWS * CDIM * 4);
  bf16_t* aoutb = (bf16_t*)alloc((size_t)M_ROWS * CDIM * 2);     // attention out bf16
  float*  x2buf = (float*) alloc((size_t)M_ROWS * CDIM * 4);     // after proj residual
  bf16_t* xn2b  = (bf16_t*)alloc((size_t)M_ROWS * CDIM * 2);     // LN2 bf16
  float*  ybuf  = (float*) alloc((size_t)M_ROWS * HIDDIM * 4);   // gelu(fc1)
  bf16_t* zb    = (bf16_t*)alloc((size_t)M_ROWS * HIDDIM * 2);   // y + gelu(conv)
  bf16_t* wqt   = (bf16_t*)alloc((size_t)CDIM * CDIM * 2);
  bf16_t* wkt   = (bf16_t*)alloc((size_t)CDIM * CDIM * 2);
  bf16_t* wvt   = (bf16_t*)alloc((size_t)CDIM * CDIM * 2);
  bf16_t* wpt   = (bf16_t*)alloc((size_t)CDIM * CDIM * 2);
  bf16_t* fc1t  = (bf16_t*)alloc((size_t)CDIM * HIDDIM * 2);
  bf16_t* fc2t  = (bf16_t*)alloc((size_t)HIDDIM * CDIM * 2);

  // ---- weight prep (bf16, N-major) ----
  int nw1 = (CDIM * CDIM + 255) / 256;
  int nw2 = (CDIM * HIDDIM + 255) / 256;
  wprep_kernel<<<nw1, 256, 0, stream>>>(wq,     wqt,  CDIM,   CDIM);
  wprep_kernel<<<nw1, 256, 0, stream>>>(wk,     wkt,  CDIM,   CDIM);
  wprep_kernel<<<nw1, 256, 0, stream>>>(wv,     wvt,  CDIM,   CDIM);
  wprep_kernel<<<nw1, 256, 0, stream>>>(proj_w, wpt,  CDIM,   CDIM);
  wprep_kernel<<<nw2, 256, 0, stream>>>(fc1_w,  fc1t, CDIM,   HIDDIM);
  wprep_kernel<<<nw2, 256, 0, stream>>>(fc2_w,  fc2t, HIDDIM, CDIM);

  // ---- LN1 + transpose ----
  ln1_transpose_kernel<<<M_ROWS / 8, 256, 0, stream>>>(feat, ln1_g, ln1_b, xbuf, xnb);

  // ---- Q, K, V projections (WMMA) ----
  const float scaleq = 0.08838834764831845f;  // 128^-0.5
  dim3 g128(M_ROWS / 16, CDIM / 64), g256(M_ROWS / 16, HIDDIM / 64);
  gemm_wmma_kernel<<<g128, 128, 0, stream>>>(xnb, wqt, bq, nullptr, qbuf, CDIM, CDIM, scaleq, 0);
  gemm_wmma_kernel<<<g128, 128, 0, stream>>>(xnb, wkt, bk, nullptr, kbuf, CDIM, CDIM, 1.0f, 0);
  gemm_wmma_kernel<<<g128, 128, 0, stream>>>(xnb, wvt, bv, nullptr, vbuf, CDIM, CDIM, 1.0f, 0);

  // ---- gather attention ----
  attn_kernel<<<M_ROWS / 8, 256, 0, stream>>>(qbuf, kbuf, vbuf, widx, dp, k_pe, v_pe, rpe, aoutb);

  // ---- proj + residual ----
  gemm_wmma_kernel<<<g128, 128, 0, stream>>>(aoutb, wpt, proj_b, xbuf, x2buf, CDIM, CDIM, 1.0f, 0);

  // ---- LN2 ----
  ln_rows_kernel<<<M_ROWS / 8, 256, 0, stream>>>(x2buf, ln2_g, ln2_b, xn2b);

  // ---- fc1 + GELU ----
  gemm_wmma_kernel<<<g256, 128, 0, stream>>>(xn2b, fc1t, fc1_b, nullptr, ybuf, CDIM, HIDDIM, 1.0f, 1);

  // ---- depthwise conv + GELU + add ----
  dwconv_kernel<<<M_ROWS, HIDDIM, 0, stream>>>(ybuf, dw_w, dw_b, zb);

  // ---- fc2 + residual -> output ----
  gemm_wmma_kernel<<<g128, 128, 0, stream>>>(zb, fc2t, fc2_b, x2buf, out, HIDDIM, CDIM, 1.0f, 0);
}
